// MPNN_15161234555429
// MI455X (gfx1250) — matrix operand order repaired and checked by reading, not compiled, not run on hardware
//
#include <hip/hip_runtime.h>

// ---------------------------------------------------------------------------
// MEGNet-style MPNN forward on gfx1250 (MI455X).
//
// Heavy GEMMs ([E,384]x[384,128] etc.) run on v_wmma_f32_16x16x32_f16 with
// fp32 accumulate. 64-row blocks: each wave owns one 16-col tile, holds the
// per-k-step B fragment (pre-packed into exact per-lane WMMA-B order) in
// registers and reuses it across 4 M-subtiles -> 4 WMMAs per B load.
// Intermediate activations and gather mirrors are stored f16 (next consumer
// is an f16 WMMA); residual masters, BN reductions and scatter-means stay
// fp32. BN-apply + residual + scatter-add are fused into one pass.
// Store epilogue: uniform out-pointer tests hoisted outside fully-unrolled
// loops; constant-offset addressing so stores clause with immediate offsets.
//
// Input pointer order: assumes the harness flattens setup_inputs() with JAX
// tree_flatten (dict keys sorted alphabetically, lists in order):
//   [0] edge_fea [E*64] f32     [1] idx1 [E] i32   [2] idx2 [E] i32
//   [3] idx3 [N] i32            [4] node_fea [N] i32
//   [5] conv_to_fc.b [256]      [6] conv_to_fc.w [256*256]
//   [7 + 16*c + k] convs[c], k: 0 bn1.b, 1 bn1.g, 2 bn2.b, 3 bn2.g,
//        4 pe0.b, 5 pe0.w[384*128], 6 pe1.b, 7 pe1.w[128*128], 8 pe2.b,
//        9 pe2.w[128*128], 10 pv0.b, 11 pv0.w[256*128], 12 pv1.b,
//        13 pv1.w[128*128], 14 pv2.b, 15 pv2.w[128*128]
//   [55] e_emb.b [128]          [56] e_emb.w [64*128]
//   [57] fc_out.b [2]           [58] fc_out.w [256*2]
//   [59] fcs0.b [256]           [60] fcs0.w [256*256]
//   [61] v_emb [100*128]
// ---------------------------------------------------------------------------

typedef __attribute__((ext_vector_type(16))) _Float16 v16h;
typedef __attribute__((ext_vector_type(8)))  _Float16 v8h;
typedef __attribute__((ext_vector_type(8)))  float    v8f;

static constexpr int   kN   = 20000;
static constexpr int   kE   = 320000;
static constexpr int   kG   = 500;
static constexpr int   kDE0 = 64;
static constexpr float kNEG = 0.2f;
static constexpr float kEPS = 1e-5f;
static constexpr int   kBNBLK = 256;   // partial-reduction blocks for BN

// ---------------------------------------------------------------------------
// Weight pre-pack: fp32 W[K,128] -> f16 in per-lane WMMA-B order.
// packed[(((kt*8+nt)*32+lane)*16)+h] = W[kt*32 + (lane>=16?16:0) + h][nt*16 + lane%16]
// ---------------------------------------------------------------------------
__global__ void pack_w_kernel(const float* __restrict__ W,
                              _Float16* __restrict__ out, int K) {
  int gid = blockIdx.x * 256 + threadIdx.x;
  int total = K * 128;
  if (gid >= total) return;
  int h    = gid & 15;
  int rest = gid >> 4;
  int lane = rest & 31;
  rest >>= 5;
  int nt = rest & 7;
  int kt = rest >> 3;
  int k  = kt * 32 + ((lane >> 4) << 4) + h;
  int n  = nt * 16 + (lane & 15);
  out[gid] = (_Float16)W[k * 128 + n];
}

// ---------------------------------------------------------------------------
// Generic fused-gather WMMA GEMM:
//   OUT[M,128] = act(rowcat(s0,s1,s2)[M,K] @ W + b)
// Block: 256 threads = 8 waves; 16*RT rows x 128 cols per block.
// Sources may be f32 (fmt=0) or f16 (fmt=1); optional row-gather per source.
// Output written as f32 and/or f16.
// ---------------------------------------------------------------------------
template <int K, int RT>
__global__ __launch_bounds__(256) void gemm16_kernel(
    const void* __restrict__ s0, const int* __restrict__ g0, int w0, int f0,
    const void* __restrict__ s1, const int* __restrict__ g1, int w1, int f1,
    const void* __restrict__ s2, int w2, int f2,
    const _Float16* __restrict__ Wp, const float* __restrict__ bias,
    float* __restrict__ out32, _Float16* __restrict__ out16, int act) {
  constexpr int KP = K + 8;  // +8 halves: rows stay 16B-aligned, no bank conflicts
  constexpr int MT = 16 * RT;
  __shared__ __align__(16) _Float16 Als[MT * KP];

  const int  tid  = threadIdx.x;
  const long base = (long)blockIdx.x * MT;
  constexpr int halfK = K / 2;

  // Stage MT rows of the gathered/concatenated input as f16 into LDS.
  for (int i = tid; i < MT * halfK; i += 256) {
    int  r   = i / halfK;
    int  col = (i - r * halfK) * 2;
    long row = base + r;
    const void* sp; const int* gp; int wseg, loc, fmt;
    if (col < w0)           { sp = s0; gp = g0;      wseg = w0; loc = col;           fmt = f0; }
    else if (col < w0 + w1) { sp = s1; gp = g1;      wseg = w1; loc = col - w0;      fmt = f1; }
    else                    { sp = s2; gp = nullptr; wseg = w2; loc = col - w0 - w1; fmt = f2; }
    long rid = gp ? (long)gp[row] : row;
    _Float16 h0, h1;
    if (fmt) {
      const _Float16* q = (const _Float16*)sp + rid * (long)wseg + loc;
      h0 = q[0]; h1 = q[1];
    } else {
      const float* q = (const float*)sp + rid * (long)wseg + loc;
      h0 = (_Float16)q[0]; h1 = (_Float16)q[1];
    }
    Als[r * KP + col]     = h0;
    Als[r * KP + col + 1] = h1;
  }
  __syncthreads();

  const int lane = tid & 31;
  const int nt   = tid >> 5;   // wave id == N-tile (8 tiles of 16 cols)
  const int m16  = lane & 15;
  const int hi   = lane >> 4;

  v8f acc[RT] = {};
#pragma unroll
  for (int kt = 0; kt < K / 32; ++kt) {
    // B fragment: contiguous 32B per lane thanks to pre-pack; reused RT times.
    v16h b = *(const v16h*)(Wp + (((long)kt * 8 + nt) * 32 + lane) * 16);
#pragma unroll
    for (int t = 0; t < RT; ++t) {
      // A fragment k order used here (matches B pack): lanes<16 -> K {0..15}; lanes>=16 -> K {16..31}.
      const _Float16* rowp = &Als[(t * 16 + m16) * KP + kt * 32 + hi * 16];
      v8h a0 = *(const v8h*)(rowp);
      v8h a1 = *(const v8h*)(rowp + 8);
      v16h a;
#pragma unroll
      for (int h = 0; h < 8; ++h) { a[h] = a0[h]; a[h + 8] = a1[h]; }
      acc[t] = __builtin_amdgcn_wmma_f32_16x16x32_f16(
          false, a, false, b, (short)0, acc[t], false, false);
    }
  }

  // Epilogue: bias + activation into registers, then branch-hoisted stores
  // from one per-lane base with compile-time offsets (clauses, imm offsets).
  const int   col = nt * 16 + m16;
  const float bv  = bias[col];
  float vals[RT * 8];
#pragma unroll
  for (int t = 0; t < RT; ++t) {
#pragma unroll
    for (int v = 0; v < 8; ++v) {
      float val = acc[t][v] + bv;
      if (act) val = (val > 0.f) ? val : kNEG * val;
      vals[t * 8 + v] = val;
    }
  }
  const long rbase = (base + hi * 8) * 128 + col;
  if (out32) {
    float* p = out32 + rbase;
#pragma unroll
    for (int t = 0; t < RT; ++t)
#pragma unroll
      for (int v = 0; v < 8; ++v)
        p[(t * 16 + v) * 128] = vals[t * 8 + v];
  }
  if (out16) {
    _Float16* p = out16 + rbase;
#pragma unroll
    for (int t = 0; t < RT; ++t)
#pragma unroll
      for (int v = 0; v < 8; ++v)
        p[(t * 16 + v) * 128] = (_Float16)vals[t * 8 + v];
  }
}

// ---------------------------------------------------------------------------
// Small helper kernels
// ---------------------------------------------------------------------------
__global__ void embed_kernel(const int* __restrict__ nf,
                             const float* __restrict__ vemb,
                             float* __restrict__ node32,
                             _Float16* __restrict__ node16, int n) {
  long gid = (long)blockIdx.x * 256 + threadIdx.x;
  if (gid >= (long)n * 128) return;
  long r = gid >> 7;
  int  c = gid & 127;
  float v = vemb[(long)nf[r] * 128 + c];
  node32[gid] = v;
  node16[gid] = (_Float16)v;
}

__global__ void count_kernel(const int* __restrict__ idx, int* __restrict__ cnt,
                             int m) {
  int gid = blockIdx.x * 256 + threadIdx.x;
  if (gid < m) atomicAdd(&cnt[idx[gid]], 1);
}

// deterministic two-level BN reduction over f16 activations (fp32 accum)
__global__ void bn_partial_kernel(const _Float16* __restrict__ x, int M,
                                  float* __restrict__ part) {
  int  col   = threadIdx.x;  // 128 threads
  int  blk   = blockIdx.x;   // kBNBLK blocks
  int  chunk = (M + kBNBLK - 1) / kBNBLK;
  long r0 = (long)blk * chunk;
  long r1 = r0 + chunk; if (r1 > M) r1 = M;
  float s = 0.f, ss = 0.f;
  for (long r = r0; r < r1; ++r) {
    float v = (float)x[r * 128 + col];
    s += v; ss += v * v;
  }
  part[(blk * 2 + 0) * 128 + col] = s;
  part[(blk * 2 + 1) * 128 + col] = ss;
}

__global__ void bn_final_kernel(const float* __restrict__ part,
                                const float* __restrict__ g,
                                const float* __restrict__ b,
                                float* __restrict__ ac, float invM) {
  int col = threadIdx.x;  // 128 threads
  float s = 0.f, ss = 0.f;
  for (int i = 0; i < kBNBLK; ++i) {
    s  += part[(i * 2 + 0) * 128 + col];
    ss += part[(i * 2 + 1) * 128 + col];
  }
  float m   = s * invM;
  float var = ss * invM - m * m;
  if (var < 0.f) var = 0.f;
  float a = g[col] * rsqrtf(var + kEPS);
  ac[col]       = a;
  ac[128 + col] = b[col] - m * a;
}

// ek = a*x + c ; edge += ek (f32 master + f16 mirror) ; vbar[idx1] += ek
__global__ void bn_apply_scatter_kernel(const _Float16* __restrict__ x,
                                        const float* __restrict__ ac,
                                        const int* __restrict__ idx,
                                        float* __restrict__ edge32,
                                        _Float16* __restrict__ edge16,
                                        float* __restrict__ vbar, long M) {
  long gid = (long)blockIdx.x * 256 + threadIdx.x;
  if (gid >= M * 128) return;
  long r   = gid >> 7;
  int  col = gid & 127;
  float y = (float)x[gid] * ac[col] + ac[128 + col];
  float e = edge32[gid] + y;
  edge32[gid] = e;
  edge16[gid] = (_Float16)e;
  atomicAdd(&vbar[(long)idx[r] * 128 + col], y);
}

// vi = a*x + c ; node += vi (f32 master + f16 mirror)
__global__ void bn_apply_node_kernel(const _Float16* __restrict__ x,
                                     const float* __restrict__ ac,
                                     float* __restrict__ node32,
                                     _Float16* __restrict__ node16, long M) {
  long gid = (long)blockIdx.x * 256 + threadIdx.x;
  if (gid >= M * 128) return;
  int   col = gid & 127;
  float y = (float)x[gid] * ac[col] + ac[128 + col];
  float v = node32[gid] + y;
  node32[gid] = v;
  node16[gid] = (_Float16)v;
}

__global__ void scatter_add_kernel(const float* __restrict__ x,
                                   const int* __restrict__ idx,
                                   float* __restrict__ acc, long M) {
  long gid = (long)blockIdx.x * 256 + threadIdx.x;
  if (gid >= M * 128) return;
  long r = gid >> 7;
  int  c = gid & 127;
  atomicAdd(&acc[(long)idx[r] * 128 + c], x[gid]);
}

__global__ void div_rows_kernel(float* __restrict__ acc,
                                const int* __restrict__ cnt, long M, int C) {
  long gid = (long)blockIdx.x * 256 + threadIdx.x;
  if (gid >= M * C) return;
  long r = gid / C;
  int  c = cnt[r]; if (c < 1) c = 1;
  acc[gid] /= (float)c;
}

__global__ void crys_scatter_kernel(const float* __restrict__ vbar,
                                    const float* __restrict__ node,
                                    const int* __restrict__ idx3,
                                    float* __restrict__ crys,
                                    int* __restrict__ cnt3, int n) {
  long gid = (long)blockIdx.x * 256 + threadIdx.x;
  if (gid >= (long)n * 256) return;
  long r = gid >> 8;
  int  c = gid & 255;
  float v = (c < 128) ? vbar[r * 128 + c] : node[r * 128 + (c - 128)];
  int  g = idx3[r];
  atomicAdd(&crys[(long)g * 256 + c], v);
  if (c == 0) atomicAdd(&cnt3[g], 1);
}

__global__ void fc_kernel(const float* __restrict__ x,
                          const float* __restrict__ w,
                          const float* __restrict__ b, float* __restrict__ y,
                          int M, int K, int Nout, int act) {
  long gid = (long)blockIdx.x * 256 + threadIdx.x;
  if (gid >= (long)M * Nout) return;
  int m = (int)(gid / Nout);
  int n = (int)(gid - (long)m * Nout);
  float acc = b[n];
  for (int k = 0; k < K; ++k) acc += x[(long)m * K + k] * w[(long)k * Nout + n];
  if (act) acc = (acc > 0.f) ? acc : kNEG * acc;
  y[gid] = acc;
}

// ---------------------------------------------------------------------------
// Host launcher
// ---------------------------------------------------------------------------
static inline size_t au(size_t x) { return (x + 255) & ~size_t(255); }

extern "C" void kernel_launch(void* const* d_in, const int* in_sizes, int n_in,
                              void* d_out, int out_size, void* d_ws,
                              size_t ws_size, hipStream_t stream) {
  (void)in_sizes; (void)n_in; (void)out_size; (void)ws_size;

  const float* edge_fea = (const float*)d_in[0];
  const int*   idx1     = (const int*)d_in[1];
  const int*   idx2     = (const int*)d_in[2];
  const int*   idx3     = (const int*)d_in[3];
  const int*   node_fea = (const int*)d_in[4];
  const float* ctf_b    = (const float*)d_in[5];
  const float* ctf_w    = (const float*)d_in[6];
  const float* eemb_b   = (const float*)d_in[55];
  const float* eemb_w   = (const float*)d_in[56];
  const float* fco_b    = (const float*)d_in[57];
  const float* fco_w    = (const float*)d_in[58];
  const float* fcs_b    = (const float*)d_in[59];
  const float* fcs_w    = (const float*)d_in[60];
  const float* v_emb    = (const float*)d_in[61];
  float*       outp     = (float*)d_out;

  // ---- workspace layout ----
  char*  ws  = (char*)d_ws;
  size_t off = 0;
  auto carve = [&](size_t bytes) { void* p = ws + off; off = au(off + bytes); return p; };
  float*    edge32  = (float*)   carve((size_t)kE * 128 * 4);
  _Float16* edge16  = (_Float16*)carve((size_t)kE * 128 * 2);
  _Float16* bufH1   = (_Float16*)carve((size_t)kE * 128 * 2);
  _Float16* bufH2   = (_Float16*)carve((size_t)kE * 128 * 2);
  _Float16* bufRaw  = (_Float16*)carve((size_t)kE * 128 * 2);
  float*    node32  = (float*)   carve((size_t)kN * 128 * 4);
  _Float16* node16  = (_Float16*)carve((size_t)kN * 128 * 2);
  float*    vbar    = (float*)   carve((size_t)kN * 128 * 4);
  _Float16* nbufH1  = (_Float16*)carve((size_t)kN * 128 * 2);
  _Float16* nbufH2  = (_Float16*)carve((size_t)kN * 128 * 2);
  _Float16* nbufRaw = (_Float16*)carve((size_t)kN * 128 * 2);
  int*      cnts    = (int*)     carve((size_t)kN * 4);
  int*      cnt3    = (int*)     carve((size_t)kG * 4);
  float*    part    = (float*)   carve((size_t)kBNBLK * 2 * 128 * 4);
  float*    ac      = (float*)   carve(256 * 4);
  float*    crysA   = (float*)   carve((size_t)kG * 256 * 4);
  float*    crysB   = (float*)   carve((size_t)kG * 256 * 4);
  // packed f16 weights
  _Float16* pWe = (_Float16*)carve((size_t)kDE0 * 128 * 2);
  _Float16* pe0[3]; _Float16* pe1[3]; _Float16* pe2[3];
  _Float16* pv0[3]; _Float16* pv1[3]; _Float16* pv2[3];
  for (int c = 0; c < 3; ++c) {
    pe0[c] = (_Float16*)carve((size_t)384 * 128 * 2);
    pe1[c] = (_Float16*)carve((size_t)128 * 128 * 2);
    pe2[c] = (_Float16*)carve((size_t)128 * 128 * 2);
    pv0[c] = (_Float16*)carve((size_t)256 * 128 * 2);
    pv1[c] = (_Float16*)carve((size_t)128 * 128 * 2);
    pv2[c] = (_Float16*)carve((size_t)128 * 128 * 2);
  }

  auto gupN = [](long n) { return (unsigned)((n + 255) / 256); };

  // ---- pack weights ----
  auto pack = [&](const float* w, _Float16* dst, int K) {
    pack_w_kernel<<<gupN((long)K * 128), 256, 0, stream>>>(w, dst, K);
  };
  pack(eemb_w, pWe, 64);
  for (int c = 0; c < 3; ++c) {
    int ci = 7 + c * 16;
    pack((const float*)d_in[ci + 5],  pe0[c], 384);
    pack((const float*)d_in[ci + 7],  pe1[c], 128);
    pack((const float*)d_in[ci + 9],  pe2[c], 128);
    pack((const float*)d_in[ci + 11], pv0[c], 256);
    pack((const float*)d_in[ci + 13], pv1[c], 128);
    pack((const float*)d_in[ci + 15], pv2[c], 128);
  }

  // ---- degree counts (idx1), node embedding, edge embedding ----
  hipMemsetAsync(cnts, 0, (size_t)kN * 4, stream);
  count_kernel<<<gupN(kE), 256, 0, stream>>>(idx1, cnts, kE);
  embed_kernel<<<gupN((long)kN * 128), 256, 0, stream>>>(node_fea, v_emb,
                                                         node32, node16, kN);
  gemm16_kernel<64, 4><<<kE / 64, 256, 0, stream>>>(
      edge_fea, nullptr, 64, 0, nullptr, nullptr, 0, 0, nullptr, 0, 0,
      pWe, eemb_b, edge32, edge16, 0);

  // ---- conv layers ----
  for (int c = 0; c < 3; ++c) {
    int ci = 7 + c * 16;
    const float* bn1_b = (const float*)d_in[ci + 0];
    const float* bn1_g = (const float*)d_in[ci + 1];
    const float* bn2_b = (const float*)d_in[ci + 2];
    const float* bn2_g = (const float*)d_in[ci + 3];
    const float* be0 = (const float*)d_in[ci + 4];
    const float* be1 = (const float*)d_in[ci + 6];
    const float* be2 = (const float*)d_in[ci + 8];
    const float* bv0 = (const float*)d_in[ci + 10];
    const float* bv1 = (const float*)d_in[ci + 12];
    const float* bv2 = (const float*)d_in[ci + 14];

    // phi_e MLP on z = [node[idx1] | node[idx2] | edge]  (all-f16 gather)
    gemm16_kernel<384, 4><<<kE / 64, 256, 0, stream>>>(
        node16, idx1, 128, 1, node16, idx2, 128, 1, edge16, 128, 1,
        pe0[c], be0, nullptr, bufH1, 1);
    gemm16_kernel<128, 4><<<kE / 64, 256, 0, stream>>>(
        bufH1, nullptr, 128, 1, nullptr, nullptr, 0, 0, nullptr, 0, 0,
        pe1[c], be1, nullptr, bufH2, 1);
    gemm16_kernel<128, 4><<<kE / 64, 256, 0, stream>>>(
        bufH2, nullptr, 128, 1, nullptr, nullptr, 0, 0, nullptr, 0, 0,
        pe2[c], be2, nullptr, bufRaw, 0);
    // bn1 -> ek ; edge += ek ; vbar = scatter_mean(ek, idx1)  (fused)
    bn_partial_kernel<<<kBNBLK, 128, 0, stream>>>(bufRaw, kE, part);
    bn_final_kernel<<<1, 128, 0, stream>>>(part, bn1_g, bn1_b, ac, 1.f / kE);
    hipMemsetAsync(vbar, 0, (size_t)kN * 128 * 4, stream);
    bn_apply_scatter_kernel<<<gupN((long)kE * 128), 256, 0, stream>>>(
        bufRaw, ac, idx1, edge32, edge16, vbar, kE);
    div_rows_kernel<<<gupN((long)kN * 128), 256, 0, stream>>>(vbar, cnts, kN, 128);
    // phi_v MLP on [vbar | node]
    gemm16_kernel<256, 2><<<kN / 32, 256, 0, stream>>>(
        vbar, nullptr, 128, 0, node16, nullptr, 128, 1, nullptr, 0, 0,
        pv0[c], bv0, nullptr, nbufH1, 1);
    gemm16_kernel<128, 2><<<kN / 32, 256, 0, stream>>>(
        nbufH1, nullptr, 128, 1, nullptr, nullptr, 0, 0, nullptr, 0, 0,
        pv1[c], bv1, nullptr, nbufH2, 1);
    gemm16_kernel<128, 2><<<kN / 32, 256, 0, stream>>>(
        nbufH2, nullptr, 128, 1, nullptr, nullptr, 0, 0, nullptr, 0, 0,
        pv2[c], bv2, nullptr, nbufRaw, 0);
    // bn2 -> node += vi
    bn_partial_kernel<<<kBNBLK, 128, 0, stream>>>(nbufRaw, kN, part);
    bn_final_kernel<<<1, 128, 0, stream>>>(part, bn2_g, bn2_b, ac, 1.f / kN);
    bn_apply_node_kernel<<<gupN((long)kN * 128), 256, 0, stream>>>(
        nbufRaw, ac, node32, node16, kN);
  }

  // ---- readout ----
  hipMemsetAsync(vbar, 0, (size_t)kN * 128 * 4, stream);
  scatter_add_kernel<<<gupN((long)kE * 128), 256, 0, stream>>>(edge32, idx1, vbar, kE);
  div_rows_kernel<<<gupN((long)kN * 128), 256, 0, stream>>>(vbar, cnts, kN, 128);

  hipMemsetAsync(crysA, 0, (size_t)kG * 256 * 4, stream);
  hipMemsetAsync(cnt3, 0, (size_t)kG * 4, stream);
  crys_scatter_kernel<<<gupN((long)kN * 256), 256, 0, stream>>>(
      vbar, node32, idx3, crysA, cnt3, kN);
  div_rows_kernel<<<gupN((long)kG * 256), 256, 0, stream>>>(crysA, cnt3, kG, 256);

  fc_kernel<<<gupN((long)kG * 256), 256, 0, stream>>>(crysA, ctf_w, ctf_b, crysB, kG, 256, 256, 1);
  fc_kernel<<<gupN((long)kG * 256), 256, 0, stream>>>(crysB, fcs_w, fcs_b, crysA, kG, 256, 256, 1);
  fc_kernel<<<gupN((long)kG * 2), 256, 0, stream>>>(crysA, fco_w, fco_b, outp, kG, 256, 2, 0);
}
